// Encoder_29197187678247
// MI455X (gfx1250) — matrix operand order repaired and checked by reading, not compiled, not run on hardware
//
#include <hip/hip_runtime.h>
#include <math.h>

// ---------------------------------------------------------------------------
// Transformer encoder forward for gfx1250 (MI455X).
// B=4, T=1024, D=1024, H=16, HD=64, L=8.
// bf16 WMMA (v_wmma_f32_16x16x32_bf16) everywhere; weights converted AND
// TRANSPOSED f32->bf16 once into workspace (96MB, L2-resident on the 192MB
// global L2). GEMM tiles are moved global->LDS by the CDNA5 Tensor Data
// Mover (tensor_load_to_lds, TENSORcnt-tracked, double-buffered), with the
// TDM pad feature reproducing the bank-conflict-free 80B LDS row stride.
// Every WMMA fragment is a pair of contiguous ds_load_b128. The K-loop is
// kept at unroll 1 with pointer-based buffer toggling so accumulators stay
// pinned in their registers (no inter-iteration v_mov copies).
// Residual stream h lives directly in d_out (fp32).
// ---------------------------------------------------------------------------

typedef __attribute__((ext_vector_type(16))) __bf16 v16bf;
typedef __attribute__((ext_vector_type(8)))  float  v8f;
typedef __attribute__((ext_vector_type(4)))  unsigned int v4u;
typedef __attribute__((ext_vector_type(8)))  int    v8i;
typedef __attribute__((ext_vector_type(4)))  int    v4i;
typedef unsigned short ushort_t;

constexpr int Bc = 4, Tc = 1024, Dc = 1024, Hc = 16, Lc = 8;
constexpr int HDc = Dc / Hc;            // 64
constexpr int Mc = Bc * Tc;             // 4096 activation rows
constexpr float EPSc = 1e-5f;
constexpr float SCALEc = 0.03125f;      // 1/sqrt(D) = 1/32

__device__ __forceinline__ ushort_t f2bf(float f) {
  union { float f; unsigned u; } a; a.f = f;
  unsigned r = a.u + 0x7fffu + ((a.u >> 16) & 1u);   // RNE
  return (ushort_t)(r >> 16);
}

union FragBF { v16bf v; ushort_t u[16]; uint4 q[2]; };
union Pack8  { uint4 q; ushort_t s[8]; };

// ------------------- Tensor Data Mover: 2D tile -> LDS ---------------------
// Loads tile_rows x 32 bf16 elements (row stride = 1024 elements in global)
// into LDS at lds_off, padding each 64B row with 16B (pad_interval=3 -> every
// 16 DWORDs, pad_amount=3 -> 4 DWORDs) to produce the 40-half padded rows.
// D# per CDNA5 ISA; groups 2/3 unused (2D tile).
__device__ __forceinline__ void tdm_load_tile_2d(unsigned lds_off,
                                                 const ushort_t* gsrc,
                                                 int tile_rows) {
  unsigned long long ga = (unsigned long long)(size_t)gsrc;
  v4u g0;
  g0[0] = 1u;                                            // count=1, user mode
  g0[1] = lds_off;                                       // lds_addr
  g0[2] = (unsigned)ga;                                  // global_addr[31:0]
  g0[3] = (unsigned)((ga >> 32) & 0x01ffffffull)         // global_addr[56:32]
          | (2u << 30);                                  // type=2 ("image")
  v8i g1;
  g1[0] = (int)((1u << 16)      // data_size = 1 -> 2 bytes
              | (1u << 20)      // pad_enable
              | (3u << 22)      // pad_interval: 16 DWORDs
              | (3u << 25));    // pad_amount: 4 DWORDs
  g1[1] = (int)(32u << 16);                              // tensor_dim0 = 32
  g1[2] = (int)((unsigned)tile_rows << 16);              // tensor_dim1 lo
  g1[3] = (int)(32u << 16);                              // tile_dim0 = 32
  g1[4] = tile_rows;                                     // tile_dim1
  g1[5] = (int)Dc;                                       // tensor_dim0_stride
  g1[6] = 0;
  g1[7] = 0;
  v4i z4 = {0, 0, 0, 0};
#if defined(__clang_major__) && (__clang_major__ >= 23)
  v8i z8 = {0, 0, 0, 0, 0, 0, 0, 0};
  __builtin_amdgcn_tensor_load_to_lds(g0, g1, z4, z4, z8, 0);
#else
  __builtin_amdgcn_tensor_load_to_lds(g0, g1, z4, z4, 0);
#endif
}

// --------------------------- embedding -------------------------------------
__global__ void embed_kernel(const int* __restrict__ x,
                             const float* __restrict__ tok,
                             float* __restrict__ h) {
  int row = blockIdx.x;                 // = b*T + t
  int t = row % Tc;
  int tokid = x[row];
  const float LOG1E4 = 9.210340371976184f;
  for (int d = threadIdx.x; d < Dc; d += blockDim.x) {
    int i2 = d & ~1;
    float inv = __expf(-((float)i2 / (float)Dc) * LOG1E4);
    float ang = (float)t * inv;
    float pe = (d & 1) ? __cosf(ang) : __sinf(ang);
    h[(size_t)row * Dc + d] = tok[(size_t)tokid * Dc + d] + pe;
  }
}

// ---------------- f32 [K][N] -> bf16 transposed [N][K] ---------------------
__global__ __launch_bounds__(256)
void cvt_t_kernel(const float* __restrict__ in, ushort_t* __restrict__ out) {
  __shared__ ushort_t tile[32][33];
  const float* src = in + (size_t)blockIdx.z * Dc * Dc;
  ushort_t* dst = out + (size_t)blockIdx.z * Dc * Dc;
  int k0 = blockIdx.y * 32, n0 = blockIdx.x * 32;
  int tx = threadIdx.x & 31, ty = threadIdx.x >> 5;     // 32 x 8
#pragma unroll
  for (int r = 0; r < 32; r += 8)
    tile[ty + r][tx] = f2bf(src[(size_t)(k0 + ty + r) * Dc + n0 + tx]);
  __syncthreads();
#pragma unroll
  for (int r = 0; r < 32; r += 8)
    dst[(size_t)(n0 + ty + r) * Dc + k0 + tx] = tile[tx][ty + r];
}

// --------------------------- layernorm -------------------------------------
__global__ __launch_bounds__(256)
void ln_kernel(const float* __restrict__ h,
               const float* __restrict__ sc,
               const float* __restrict__ bi,
               ushort_t* __restrict__ out) {
  int row = blockIdx.x;
  const float* xr = h + (size_t)row * Dc;
  float vals[4];
  float lsum = 0.f, lsq = 0.f;
#pragma unroll
  for (int i = 0; i < 4; ++i) {
    float v = xr[threadIdx.x + i * 256];
    vals[i] = v; lsum += v; lsq += v * v;
  }
#pragma unroll
  for (int m = 1; m < 32; m <<= 1) {
    lsum += __shfl_xor(lsum, m, 32);
    lsq  += __shfl_xor(lsq,  m, 32);
  }
  __shared__ float ssum[8], ssq[8];
  int w = threadIdx.x >> 5;
  if ((threadIdx.x & 31) == 0) { ssum[w] = lsum; ssq[w] = lsq; }
  __syncthreads();
  float tsum = 0.f, tsq = 0.f;
#pragma unroll
  for (int i = 0; i < 8; ++i) { tsum += ssum[i]; tsq += ssq[i]; }
  float mu  = tsum * (1.0f / Dc);
  float var = tsq * (1.0f / Dc) - mu * mu;
  float rr  = rsqrtf(var + EPSc);
#pragma unroll
  for (int i = 0; i < 4; ++i) {
    int d = threadIdx.x + i * 256;
    float y = (vals[i] - mu) * rr * sc[d] + bi[d];
    out[(size_t)row * Dc + d] = f2bf(y);
  }
}

// --------------------------- GEMM (bf16 WMMA + TDM) ------------------------
// C[M,1024] = A[M,1024] @ W[1024,1024] + bias, W pre-transposed [N][K].
// Block tile 128x128, K-step 32; 8 waves, each wave 32x64 (2x4 WMMA tiles).
// Double-buffered LDS via pointer toggle; TDM prefetch issued from wave 0.
constexpr int BMg = 128, BNg = 128, BKg = 32;
constexpr int LDP = 40;                 // padded LDS row stride (halves)
enum { MODE_BF16 = 0, MODE_RELU = 1, MODE_RESID = 2 };

__global__ __launch_bounds__(256)
void gemm_kernel(const ushort_t* __restrict__ A,
                 const ushort_t* __restrict__ Wt,   // [N][K] bf16
                 const float* __restrict__ bias,
                 ushort_t* __restrict__ outB,
                 float* __restrict__ outF,
                 int mode) {
  __shared__ ushort_t As[2][BMg][LDP];
  __shared__ ushort_t Bs[2][BNg][LDP];
  const int tid  = threadIdx.x;
  const int m0g  = blockIdx.y * BMg;
  const int n0g  = blockIdx.x * BNg;
  const int wave = tid >> 5, lane = tid & 31;
  const int waveM = (wave >> 1) * 32;
  const int waveN = (wave & 1) * 64;
  const int ml = lane & 15, mh = lane >> 4;

  const ushort_t* Abase = A  + (size_t)m0g * Dc;
  const ushort_t* Bbase = Wt + (size_t)n0g * Dc;
  unsigned aOff[2] = { (unsigned)(size_t)&As[0][0][0], (unsigned)(size_t)&As[1][0][0] };
  unsigned bOff[2] = { (unsigned)(size_t)&Bs[0][0][0], (unsigned)(size_t)&Bs[1][0][0] };

  if (wave == 0) {
    tdm_load_tile_2d(aOff[0], Abase, BMg);
    tdm_load_tile_2d(bOff[0], Bbase, BNg);
    __builtin_amdgcn_s_wait_tensorcnt(0);
  }
  __syncthreads();

  // Loop-invariant per-lane fragment bases (element offsets into a buffer).
  const int aRow0 = (waveM + ml) * LDP + mh * 8;        // A frag, im-tile adds 16*LDP
  const int bRow0 = (waveN + ml) * LDP + mh * 8;        // B frag, jn-tile adds 16*LDP
  const ushort_t* a0 = &As[0][0][0];
  const ushort_t* b0 = &Bs[0][0][0];
  const int bufA = BMg * LDP, bufB = BNg * LDP;

  v8f acc[2][4] = {};
  const int NK = Dc / BKg;              // 32 K-steps
#pragma unroll 1
  for (int ki = 0; ki < NK; ++ki) {
    const ushort_t* Ac = a0 + (ki & 1) * bufA;
    const ushort_t* Bc = b0 + (ki & 1) * bufB;
    if (wave == 0 && ki + 1 < NK) {     // prefetch next tiles via TDM
      int k1 = (ki + 1) * BKg;
      tdm_load_tile_2d(aOff[(ki + 1) & 1], Abase + k1, BMg);
      tdm_load_tile_2d(bOff[(ki + 1) & 1], Bbase + k1, BNg);
    }

    FragBF aF[2], bF[4];
#pragma unroll
    for (int im = 0; im < 2; ++im) {
      const ushort_t* p = Ac + aRow0 + im * 16 * LDP;
      aF[im].q[0] = *(const uint4*)(p);
      aF[im].q[1] = *(const uint4*)(p + 16);
    }
#pragma unroll
    for (int jn = 0; jn < 4; ++jn) {
      const ushort_t* p = Bc + bRow0 + jn * 16 * LDP;
      bF[jn].q[0] = *(const uint4*)(p);
      bF[jn].q[1] = *(const uint4*)(p + 16);
    }
#pragma unroll
    for (int im = 0; im < 2; ++im)
#pragma unroll
      for (int jn = 0; jn < 4; ++jn)
        acc[im][jn] = __builtin_amdgcn_wmma_f32_16x16x32_bf16(
            false, aF[im].v, false, bF[jn].v, (short)0, acc[im][jn], false, false);

    if (wave == 0) __builtin_amdgcn_s_wait_tensorcnt(0);
    __syncthreads();                    // publish next buffer / retire reads
  }

#pragma unroll
  for (int im = 0; im < 2; ++im) {
#pragma unroll
    for (int jn = 0; jn < 4; ++jn) {
      int col = n0g + waveN + jn * 16 + ml;
      float bv = bias[col];
#pragma unroll
      for (int r = 0; r < 8; ++r) {
        int row = m0g + waveM + im * 16 + mh * 8 + r;   // C/D: M = r + 8*(l>>4)
        float vv = acc[im][jn][r] + bv;
        size_t idx = (size_t)row * Dc + col;
        if (mode == MODE_RESID)      outF[idx] += vv;
        else if (mode == MODE_RELU)  outB[idx] = f2bf(fmaxf(vv, 0.f));
        else                         outB[idx] = f2bf(vv);
      }
    }
  }
}

// --------------------------- attention (flash-style) -----------------------
__global__ __launch_bounds__(128)
void attn_kernel(const ushort_t* __restrict__ Qp,
                 const ushort_t* __restrict__ Kp,
                 const ushort_t* __restrict__ Vp,
                 ushort_t* __restrict__ Op) {
  __shared__ ushort_t Kr[32][72];        // [key][d], 144B stride
  __shared__ ushort_t Vt[HDc][LDP];      // [d][key]
  __shared__ ushort_t Ps[4][16][LDP];    // per-wave P staging
  const int tid = threadIdx.x;
  const int wave = tid >> 5, lane = tid & 31;
  const int ml = lane & 15, mh = lane >> 4;
  const int QT = Tc / 64;
  int bh = blockIdx.x / QT, qt = blockIdx.x % QT;
  int b = bh / Hc, hh = bh % Hc;
  size_t base = ((size_t)b * Tc) * Dc + (size_t)hh * HDc;
  int trow0 = qt * 64 + wave * 16;

  FragBF qF[2];
#pragma unroll
  for (int c = 0; c < 2; ++c) {
    const ushort_t* qr = Qp + base + (size_t)(trow0 + ml) * Dc + c * 32;
    qF[c].q[0] = *(const uint4*)(qr + mh * 8);
    qF[c].q[1] = *(const uint4*)(qr + 16 + mh * 8);
  }

  v8f oacc[4] = {};
  float mrun[8], lrun[8];
#pragma unroll
  for (int r = 0; r < 8; ++r) { mrun[r] = -1e30f; lrun[r] = 0.f; }

#pragma unroll 1
  for (int kt = 0; kt < Tc; kt += 32) {
#pragma unroll
    for (int i = tid; i < 256; i += 128) {
      int key = i >> 3, d8 = (i & 7) * 8;
      *(uint4*)&Kr[key][d8] =
          *(const uint4*)(Kp + base + (size_t)(kt + key) * Dc + d8);
    }
#pragma unroll
    for (int i = tid; i < 256; i += 128) {
      int key = i >> 3, d8 = (i & 7) * 8;
      Pack8 p; p.q = *(const uint4*)(Vp + base + (size_t)(kt + key) * Dc + d8);
#pragma unroll
      for (int e = 0; e < 8; ++e) Vt[d8 + e][key] = p.s[e];
    }
    __syncthreads();

    v8f sacc[2] = {};
#pragma unroll
    for (int ktile = 0; ktile < 2; ++ktile) {
      int key = ktile * 16 + ml;
#pragma unroll
      for (int c = 0; c < 2; ++c) {
        FragBF kB;
        kB.q[0] = *(const uint4*)&Kr[key][c * 32 + mh * 8];
        kB.q[1] = *(const uint4*)&Kr[key][c * 32 + 16 + mh * 8];
        sacc[ktile] = __builtin_amdgcn_wmma_f32_16x16x32_bf16(
            false, qF[c].v, false, kB.v, (short)0, sacc[ktile], false, false);
      }
    }

    float p0[8], p1[8];
#pragma unroll
    for (int r = 0; r < 8; ++r) {
      float s0 = sacc[0][r] * SCALEc;
      float s1 = sacc[1][r] * SCALEc;
      float mx = fmaxf(s0, s1);
#pragma unroll
      for (int mm = 1; mm < 16; mm <<= 1) mx = fmaxf(mx, __shfl_xor(mx, mm, 32));
      float mnew = fmaxf(mrun[r], mx);
      float e0 = __expf(s0 - mnew);
      float e1 = __expf(s1 - mnew);
      float rs = e0 + e1;
#pragma unroll
      for (int mm = 1; mm < 16; mm <<= 1) rs += __shfl_xor(rs, mm, 32);
      float fr = __expf(mrun[r] - mnew);
      lrun[r] = lrun[r] * fr + rs;
      mrun[r] = mnew;
#pragma unroll
      for (int nt = 0; nt < 4; ++nt) oacc[nt][r] *= fr;
      p0[r] = e0; p1[r] = e1;
    }

#pragma unroll
    for (int r = 0; r < 8; ++r) {
      Ps[wave][mh * 8 + r][ml]      = f2bf(p0[r]);
      Ps[wave][mh * 8 + r][16 + ml] = f2bf(p1[r]);
    }
    FragBF pF;
    pF.q[0] = *(const uint4*)&Ps[wave][ml][mh * 8];
    pF.q[1] = *(const uint4*)&Ps[wave][ml][16 + mh * 8];

#pragma unroll
    for (int nt = 0; nt < 4; ++nt) {
      FragBF vB;
      vB.q[0] = *(const uint4*)&Vt[nt * 16 + ml][mh * 8];
      vB.q[1] = *(const uint4*)&Vt[nt * 16 + ml][16 + mh * 8];
      oacc[nt] = __builtin_amdgcn_wmma_f32_16x16x32_bf16(
          false, pF.v, false, vB.v, (short)0, oacc[nt], false, false);
    }
    __syncthreads();
  }

#pragma unroll
  for (int nt = 0; nt < 4; ++nt) {
    int col = hh * HDc + nt * 16 + ml;
#pragma unroll
    for (int r = 0; r < 8; ++r) {
      int t = trow0 + mh * 8 + r;
      float vv = oacc[nt][r] / lrun[r];
      Op[((size_t)b * Tc + t) * Dc + col] = f2bf(vv);
    }
  }
}

// --------------------------- host orchestration ----------------------------
extern "C" void kernel_launch(void* const* d_in, const int* in_sizes, int n_in,
                              void* d_out, int out_size, void* d_ws, size_t ws_size,
                              hipStream_t stream) {
  const int*   x     = (const int*)  d_in[0];
  const float* tok   = (const float*)d_in[1];
  const float* ln1_s = (const float*)d_in[2];
  const float* ln1_b = (const float*)d_in[3];
  const float* Wq    = (const float*)d_in[4];
  const float* bq    = (const float*)d_in[5];
  const float* Wk    = (const float*)d_in[6];
  const float* bk    = (const float*)d_in[7];
  const float* Wv    = (const float*)d_in[8];
  const float* bv    = (const float*)d_in[9];
  const float* Wo    = (const float*)d_in[10];
  const float* bo    = (const float*)d_in[11];
  const float* ln2_s = (const float*)d_in[12];
  const float* ln2_b = (const float*)d_in[13];
  const float* W1    = (const float*)d_in[14];
  const float* b1    = (const float*)d_in[15];
  const float* W2    = (const float*)d_in[16];
  const float* b2    = (const float*)d_in[17];
  float* h = (float*)d_out;

  char* ws = (char*)d_ws;
  const size_t wsz = (size_t)Lc * Dc * Dc;
  const size_t act = (size_t)Mc * Dc;
  ushort_t* wq_b = (ushort_t*)ws;  ws += wsz * 2;
  ushort_t* wk_b = (ushort_t*)ws;  ws += wsz * 2;
  ushort_t* wv_b = (ushort_t*)ws;  ws += wsz * 2;
  ushort_t* wo_b = (ushort_t*)ws;  ws += wsz * 2;
  ushort_t* w1_b = (ushort_t*)ws;  ws += wsz * 2;
  ushort_t* w2_b = (ushort_t*)ws;  ws += wsz * 2;
  ushort_t* xn   = (ushort_t*)ws;  ws += act * 2;
  ushort_t* qb   = (ushort_t*)ws;  ws += act * 2;
  ushort_t* kb   = (ushort_t*)ws;  ws += act * 2;
  ushort_t* vb   = (ushort_t*)ws;  ws += act * 2;
  ushort_t* ao   = (ushort_t*)ws;  ws += act * 2;
  ushort_t* ff   = (ushort_t*)ws;  ws += act * 2;

  embed_kernel<<<Mc, 256, 0, stream>>>(x, tok, h);

  dim3 tgrid(Dc / 32, Dc / 32, Lc);
  cvt_t_kernel<<<tgrid, 256, 0, stream>>>(Wq, wq_b);
  cvt_t_kernel<<<tgrid, 256, 0, stream>>>(Wk, wk_b);
  cvt_t_kernel<<<tgrid, 256, 0, stream>>>(Wv, wv_b);
  cvt_t_kernel<<<tgrid, 256, 0, stream>>>(Wo, wo_b);
  cvt_t_kernel<<<tgrid, 256, 0, stream>>>(W1, w1_b);
  cvt_t_kernel<<<tgrid, 256, 0, stream>>>(W2, w2_b);

  dim3 gg(Dc / BNg, Mc / BMg);                    // (8, 32)
  const int agrid = Bc * Hc * (Tc / 64);          // 1024

  for (int l = 0; l < Lc; ++l) {
    size_t w_off = (size_t)l * Dc * Dc;
    size_t v_off = (size_t)l * Dc;
    ln_kernel<<<Mc, 256, 0, stream>>>(h, ln1_s + v_off, ln1_b + v_off, xn);
    gemm_kernel<<<gg, 256, 0, stream>>>(xn, wq_b + w_off, bq + v_off, qb, nullptr, MODE_BF16);
    gemm_kernel<<<gg, 256, 0, stream>>>(xn, wk_b + w_off, bk + v_off, kb, nullptr, MODE_BF16);
    gemm_kernel<<<gg, 256, 0, stream>>>(xn, wv_b + w_off, bv + v_off, vb, nullptr, MODE_BF16);
    attn_kernel<<<agrid, 128, 0, stream>>>(qb, kb, vb, ao);
    gemm_kernel<<<gg, 256, 0, stream>>>(ao, wo_b + w_off, bo + v_off, nullptr, h, MODE_RESID);
    ln_kernel<<<Mc, 256, 0, stream>>>(h, ln2_s + v_off, ln2_b + v_off, xn);
    gemm_kernel<<<gg, 256, 0, stream>>>(xn, w1_b + w_off, b1 + v_off, ff, nullptr, MODE_RELU);
    gemm_kernel<<<gg, 256, 0, stream>>>(ff, w2_b + w_off, b2 + v_off, nullptr, h, MODE_RESID);
  }
}
